// DisparityNetwork_44418551775458
// MI455X (gfx1250) — compile-verified, matrix operand order checked by reading, not run on hardware
//
#include <hip/hip_runtime.h>

typedef __attribute__((ext_vector_type(8)))  _Float16 v8h;
typedef __attribute__((ext_vector_type(16))) _Float16 v16h;
typedef __attribute__((ext_vector_type(8)))  float    v8f;

union HFrag { v16h v; v8h h[2]; };

#define HF 512
#define WF 1024
#define FULLPIX (4 * HF * WF)

// Padded volume layout: (B, H+2, W+2, Dd+2, 16) f16, interior at +1 in each
// spatial dim, halo permanently zero (zeroed once per scale). Channel 0 holds
// the cost for layers with cout=1; channels >= cout are zero by construction.

// ---------------------------------------------------------------------------
// Zero fill (16 bytes per thread)
// ---------------------------------------------------------------------------
__global__ void zero_kernel(uint4* __restrict__ p, long n16)
{
  long i = (long)blockIdx.x * blockDim.x + threadIdx.x;
  if (i >= n16) return;
  p[i] = make_uint4(0u, 0u, 0u, 0u);
}

// ---------------------------------------------------------------------------
// Cost volume, scale 0:  cost[b,h,w,d] = sum_c |fl[b,h,w,c] - fr[b,h,w-d,c]|
// ---------------------------------------------------------------------------
__global__ void costvol0_kernel(const float* __restrict__ fl, const float* __restrict__ fr,
                                _Float16* __restrict__ vol, int B, int H, int W, int Dd)
{
  long i = (long)blockIdx.x * blockDim.x + threadIdx.x;
  long total = (long)B * H * W * Dd;
  if (i >= total) return;
  int d = (int)(i % Dd); long r = i / Dd;
  int w = (int)(r % W);  r /= W;
  int h = (int)(r % H);  int b = (int)(r / H);

  const float* flp = fl + (((long)b * H + h) * W + w) * 32;
  float cost = 0.f;
  if (w >= d) {
    const float* frp = fr + (((long)b * H + h) * W + (w - d)) * 32;
    for (int c = 0; c < 32; ++c) cost += fabsf(flp[c] - frp[c]);
  } else {
    for (int c = 0; c < 32; ++c) cost += fabsf(flp[c]);
  }
  long pidx = ((((long)b * (H + 2) + (h + 1)) * (W + 2) + (w + 1)) * (Dd + 2) + (d + 1)) * 16;
  _Float16* vp = vol + pidx;
  vp[0] = (_Float16)cost;
#pragma unroll
  for (int c = 1; c < 16; ++c) vp[c] = (_Float16)0.f;
}

// ---------------------------------------------------------------------------
// Residual cost volume (scales 1,2): bilinear warp of fr at x = w - disp + shift
// ---------------------------------------------------------------------------
__global__ void costvol_warp_kernel(const float* __restrict__ fl, const float* __restrict__ fr,
                                    const float* __restrict__ wflow,
                                    _Float16* __restrict__ vol, int B, int H, int W, int Dd)
{
  long i = (long)blockIdx.x * blockDim.x + threadIdx.x;
  long total = (long)B * H * W * Dd;
  if (i >= total) return;
  int d = (int)(i % Dd); long r = i / Dd;
  int w = (int)(r % W);  r /= W;
  int h = (int)(r % H);  int b = (int)(r / H);

  float disp  = wflow[((long)b * H + h) * W + w];
  float shift = (float)(d - (Dd / 2));
  float xq = (float)w - disp + shift;
  xq = fminf(fmaxf(xq, 0.f), (float)(W - 1));
  int   x0 = (int)xq;
  float fr8 = xq - (float)x0;
  int   x1 = min(x0 + 1, W - 1);

  const float* flp = fl + (((long)b * H + h) * W + w)  * 32;
  const float* fp0 = fr + (((long)b * H + h) * W + x0) * 32;
  const float* fp1 = fr + (((long)b * H + h) * W + x1) * 32;
  float cost = 0.f;
  for (int c = 0; c < 32; ++c) {
    float wv = (1.f - fr8) * fp0[c] + fr8 * fp1[c];
    cost += fabsf(flp[c] - wv);
  }
  long pidx = ((((long)b * (H + 2) + (h + 1)) * (W + 2) + (w + 1)) * (Dd + 2) + (d + 1)) * 16;
  _Float16* vp = vol + pidx;
  vp[0] = (_Float16)cost;
#pragma unroll
  for (int c = 1; c < 16; ++c) vp[c] = (_Float16)0.f;
}

// ---------------------------------------------------------------------------
// Weight prep: (3,3,3,cin,cout) f32 -> wT[28][16][16] f16  (wT[tap][cout][cin]),
// zero-padded in cin/cout and tap 27; bias padded to 16 f32.
// ---------------------------------------------------------------------------
__global__ void prep_weights_kernel(const float* __restrict__ w, const float* __restrict__ bsrc,
                                    _Float16* __restrict__ wT, float* __restrict__ bias16,
                                    int cin, int cout)
{
  int i = blockIdx.x * blockDim.x + threadIdx.x;
  if (i >= 28 * 16 * 16) return;
  int ci  = i & 15;
  int co  = (i >> 4) & 15;
  int tap = i >> 8;
  float v = 0.f;
  if (tap < 27 && ci < cin && co < cout) v = w[((long)tap * cin + ci) * cout + co];
  wT[((long)tap * 16 + co) * 16 + ci] = (_Float16)v;
  if (i < 16) bias16[i] = (i < cout) ? bsrc[i] : 0.f;
}

// ---------------------------------------------------------------------------
// Conv3d 3x3x3 SAME, padded cin=16 -> cout=16, implicit GEMM via WMMA.
// One wave per 4 tiles; tile = 16 voxels along W at fixed (b,h,disp).
// Halo-padded input: no bounds checks, all tap offsets are immediates.
// Weights staged in LDS once per workgroup; B fragments read via ds_load.
// sched_barrier pins all 28 A loads ahead of the WMMA chain so loads drain
// with partial s_wait_loadcnt while the matrix pipe runs.
// ---------------------------------------------------------------------------
template <int H, int W, int Dd>
__global__ __launch_bounds__(128, 1)
void conv3d_wmma_kernel(const _Float16* __restrict__ vin, _Float16* __restrict__ vout,
                        const _Float16* __restrict__ wT, const float* __restrict__ bias16,
                        long nTiles, int relu)
{
  constexpr int  WC = W / 16;
  constexpr long SD = 16;                       // halves per disparity step
  constexpr long SW = (long)(Dd + 2) * 16;      // halves per W step
  constexpr long SH = (long)(W + 2) * SW;       // halves per H step
  constexpr long SB = (long)(H + 2) * SH;       // halves per batch step

  // ---- stage weights into LDS (once per workgroup, before any divergence) --
  __shared__ _Float16 sW[28 * 16 * 16];         // 14336 bytes
  {
    const uint4* src = (const uint4*)wT;
    uint4*       dst = (uint4*)sW;
#pragma unroll
    for (int it = 0; it < 7; ++it) {            // 896 uint4 / 128 threads
      int i = threadIdx.x + it * 128;
      dst[i] = src[i];
    }
  }
  __syncthreads();

  const int lane = threadIdx.x & 31;
  const int m    = lane & 15;                   // A row / B col / C col
  const int ch8  = (lane >> 4) << 3;            // cin half this lane holds (0 or 8)
  const float bv = bias16[m];

  const long t0 = ((long)blockIdx.x * 4 + (threadIdx.x >> 5)) * 4;
  if (t0 >= nTiles) return;                     // wave-uniform

  for (int k = 0; k < 4; ++k) {
    const long tile = t0 + k;
    if (tile >= nTiles) break;                  // wave-uniform
    int  wc  = (int)(tile % WC); long r = tile / WC;
    int  dsp = (int)(r % Dd); r /= Dd;
    int  h   = (int)(r % H);
    int  b   = (int)(r / H);
    const int w0 = wc * 16;

    const _Float16* abase = vin + (long)b * SB + (long)(h + 1) * SH
                          + (long)(w0 + 1 + m) * SW + (long)(dsp + 1) * SD + ch8;

    // phase 1: issue all 28 A-fragment loads (immediate offsets, one clause)
    HFrag a[14];
#pragma unroll
    for (int p = 0; p < 14; ++p) {
#pragma unroll
      for (int half = 0; half < 2; ++half) {
        const int t = 2 * p + half;
        v8h av = {};
        if (t < 27) {                           // compile-time: tap 27 stays zero
          const int  kd  = t / 9, kh = (t / 3) % 3, kw = t % 3;
          const long off = (long)(kd - 1) * SH + (long)(kh - 1) * SW + (long)(kw - 1) * SD;
          av = *(const v8h*)(abase + off);
        }
        a[p].h[half] = av;
      }
    }

    // hard scheduling fence: no instruction may cross -> loads stay issued
    __builtin_amdgcn_sched_barrier(0);

    // phase 2: 14 WMMAs; B fragments streamed from LDS
    v8f acc = {};
#pragma unroll
    for (int p = 0; p < 14; ++p) {
      HFrag bfr;
#pragma unroll
      for (int half = 0; half < 2; ++half) {
        const int t = 2 * p + half;
        bfr.h[half] = *(const v8h*)(sW + ((t * 16 + m) * 16 + ch8));
      }
      acc = __builtin_amdgcn_wmma_f32_16x16x32_f16(
          false, a[p].v, false, bfr.v, (short)0, acc, false, false);
    }

    // C layout: vgpr j, lane l -> M = j + (l>=16 ? 8 : 0), N = l & 15
    _Float16* wbp = vout + (long)b * SB + (long)(h + 1) * SH
                  + (long)(w0 + 1 + ch8) * SW + (long)(dsp + 1) * SD + m;
#pragma unroll
    for (int j = 0; j < 8; ++j) {
      float val = acc[j] + bv;
      if (relu) val = fmaxf(val, 0.f);
      wbp[(long)j * SW] = (_Float16)val;        // immediate-offset b16 store
    }
  }
}

// ---------------------------------------------------------------------------
// softmax(-cost) over disparity + regression
// ---------------------------------------------------------------------------
__global__ void softmax_reg_kernel(const _Float16* __restrict__ vol, float* __restrict__ plow,
                                   int B, int H, int W, int Dd, float start)
{
  long i = (long)blockIdx.x * blockDim.x + threadIdx.x;
  long total = (long)B * H * W;
  if (i >= total) return;
  int w = (int)(i % W); long r = i / W;
  int h = (int)(r % H); int b = (int)(r / H);
  const _Float16* vp = vol
      + ((((long)b * (H + 2) + (h + 1)) * (W + 2) + (w + 1)) * (Dd + 2) + 1) * 16;
  float mx = -1e30f;
  for (int d = 0; d < Dd; ++d) {
    float c = -(float)vp[d * 16];
    mx = fmaxf(mx, c);
  }
  float sum = 0.f, acc = 0.f;
  for (int d = 0; d < Dd; ++d) {
    float e = __expf(-(float)vp[d * 16] - mx);
    sum += e;
    acc += e * (start + (float)d);
  }
  plow[i] = acc / sum;
}

// ---------------------------------------------------------------------------
// Bilinear upsample (half-pixel centers, clamped) to 512x1024, optional residual
// ---------------------------------------------------------------------------
__global__ void upsample_add_kernel(const float* __restrict__ plow, int fh, int fw,
                                    const float* __restrict__ prev, float* __restrict__ outp,
                                    int B)
{
  long i = (long)blockIdx.x * blockDim.x + threadIdx.x;
  long total = (long)B * HF * WF;
  if (i >= total) return;
  int x = (int)(i % WF);
  int y = (int)((i / WF) % HF);
  int b = (int)(i / ((long)WF * HF));
  float sy = (y + 0.5f) * ((float)fh / (float)HF) - 0.5f;
  float sx = (x + 0.5f) * ((float)fw / (float)WF) - 0.5f;
  sy = fminf(fmaxf(sy, 0.f), (float)(fh - 1));
  sx = fminf(fmaxf(sx, 0.f), (float)(fw - 1));
  int y0 = (int)sy, x0 = (int)sx;
  int y1 = min(y0 + 1, fh - 1), x1 = min(x0 + 1, fw - 1);
  float fy = sy - y0, fx = sx - x0;
  const float* pb = plow + (long)b * fh * fw;
  float v00 = pb[(long)y0 * fw + x0], v01 = pb[(long)y0 * fw + x1];
  float v10 = pb[(long)y1 * fw + x0], v11 = pb[(long)y1 * fw + x1];
  float val = (1.f - fy) * ((1.f - fx) * v00 + fx * v01)
            +        fy  * ((1.f - fx) * v10 + fx * v11);
  if (prev) val += prev[i];
  outp[i] = val;
}

// ---------------------------------------------------------------------------
// Bilinear downsample of full-res prediction to (fh,fw), scaled by fh/512
// ---------------------------------------------------------------------------
__global__ void downsample_flow_kernel(const float* __restrict__ full, float* __restrict__ wflow,
                                       int B, int fh, int fw, float mul)
{
  long i = (long)blockIdx.x * blockDim.x + threadIdx.x;
  long total = (long)B * fh * fw;
  if (i >= total) return;
  int x = (int)(i % fw);
  int y = (int)((i / fw) % fh);
  int b = (int)(i / ((long)fw * fh));
  float sy = (y + 0.5f) * ((float)HF / (float)fh) - 0.5f;
  float sx = (x + 0.5f) * ((float)WF / (float)fw) - 0.5f;
  sy = fminf(fmaxf(sy, 0.f), (float)(HF - 1));
  sx = fminf(fmaxf(sx, 0.f), (float)(WF - 1));
  int y0 = (int)sy, x0 = (int)sx;
  int y1 = min(y0 + 1, HF - 1), x1 = min(x0 + 1, WF - 1);
  float fy = sy - y0, fx = sx - x0;
  const float* pb = full + (long)b * HF * WF;
  float v00 = pb[(long)y0 * WF + x0], v01 = pb[(long)y0 * WF + x1];
  float v10 = pb[(long)y1 * WF + x0], v11 = pb[(long)y1 * WF + x1];
  float val = (1.f - fy) * ((1.f - fx) * v00 + fx * v01)
            +        fy  * ((1.f - fx) * v10 + fx * v11);
  wflow[i] = val * mul;
}

// ---------------------------------------------------------------------------
extern "C" void kernel_launch(void* const* d_in, const int* in_sizes, int n_in,
                              void* d_out, int out_size, void* d_ws, size_t ws_size,
                              hipStream_t stream)
{
  (void)in_sizes; (void)n_in; (void)out_size; (void)ws_size;

  const float* fl[3] = { (const float*)d_in[0], (const float*)d_in[2], (const float*)d_in[4] };
  const float* fr[3] = { (const float*)d_in[1], (const float*)d_in[3], (const float*)d_in[5] };
  // params flattened after the 6 feature maps: [s][l] -> w at 6+s*8+l*2, b at +1

  const int B = 4;
  const int Hs[3]  = { 32, 64, 128 };
  const int Ws[3]  = { 64, 128, 256 };
  const int Dds[3] = { 12, 5, 5 };

  // workspace carve (256-byte aligned)
  char* ws = (char*)d_ws;
  size_t off = 0;
  auto carve = [&](size_t bytes) -> void* {
    void* p = ws + off;
    off += (bytes + 255) & ~(size_t)255;
    return p;
  };
  // max padded volume: scale 2 -> 4 * 130 * 258 * 7 * 16 halves
  const size_t maxPadHalves = (size_t)B * 130 * 258 * 7 * 16;
  _Float16* volA   = (_Float16*)carve(maxPadHalves * sizeof(_Float16));
  _Float16* volB   = (_Float16*)carve(maxPadHalves * sizeof(_Float16));
  _Float16* wT     = (_Float16*)carve(28 * 16 * 16 * sizeof(_Float16));
  float*    bias16 = (float*)carve(16 * sizeof(float));
  float*    plow   = (float*)carve((size_t)B * 128 * 256 * sizeof(float));
  float*    wflow  = (float*)carve((size_t)B * 128 * 256 * sizeof(float));
  float*    out    = (float*)d_out;

  for (int s = 0; s < 3; ++s) {
    const int H = Hs[s], W = Ws[s], Dd = Dds[s];
    const long nvox  = (long)B * H * W * Dd;
    const long padH  = (long)B * (H + 2) * (W + 2) * (Dd + 2) * 16;  // halves

    // 0) zero padded ping-pong volumes (halo must be zero for SAME padding)
    {
      long n16 = padH / 8;  // uint4 = 8 halves
      zero_kernel<<<(unsigned)((n16 + 255) / 256), 256, 0, stream>>>((uint4*)volA, n16);
      zero_kernel<<<(unsigned)((n16 + 255) / 256), 256, 0, stream>>>((uint4*)volB, n16);
    }

    // 1) cost volume into volA interior
    if (s == 0) {
      costvol0_kernel<<<(unsigned)((nvox + 255) / 256), 256, 0, stream>>>(
          fl[0], fr[0], volA, B, H, W, Dd);
    } else {
      long np = (long)B * H * W;
      downsample_flow_kernel<<<(unsigned)((np + 255) / 256), 256, 0, stream>>>(
          out + (long)(s - 1) * FULLPIX, wflow, B, H, W, (float)H / (float)HF);
      costvol_warp_kernel<<<(unsigned)((nvox + 255) / 256), 256, 0, stream>>>(
          fl[s], fr[s], wflow, volA, B, H, W, Dd);
    }

    // 2) regularization: 4 conv3d layers, WMMA implicit GEMM (channels padded to 16)
    const int F = (s == 0) ? 16 : 4;
    const int chans[5] = { 1, F, F, F, 1 };
    _Float16* cur = volA;
    _Float16* nxt = volB;
    const long nTiles = (long)B * H * Dd * (W / 16);
    const unsigned cblocks = (unsigned)((nTiles + 15) / 16);  // 4 waves * 4 tiles/wave
    for (int l = 0; l < 4; ++l) {
      const float* wsrc = (const float*)d_in[6 + s * 8 + l * 2];
      const float* bsrc = (const float*)d_in[6 + s * 8 + l * 2 + 1];
      prep_weights_kernel<<<(28 * 16 * 16 + 255) / 256, 256, 0, stream>>>(
          wsrc, bsrc, wT, bias16, chans[l], chans[l + 1]);
      const int relu = (l < 3) ? 1 : 0;
      if (s == 0)
        conv3d_wmma_kernel<32, 64, 12><<<cblocks, 128, 0, stream>>>(
            cur, nxt, wT, bias16, nTiles, relu);
      else if (s == 1)
        conv3d_wmma_kernel<64, 128, 5><<<cblocks, 128, 0, stream>>>(
            cur, nxt, wT, bias16, nTiles, relu);
      else
        conv3d_wmma_kernel<128, 256, 5><<<cblocks, 128, 0, stream>>>(
            cur, nxt, wT, bias16, nTiles, relu);
      _Float16* t = cur; cur = nxt; nxt = t;
    }

    // 3) softmax + disparity regression
    {
      long np = (long)B * H * W;
      float start = (s == 0) ? 0.f : -(float)(Dd / 2);
      softmax_reg_kernel<<<(unsigned)((np + 255) / 256), 256, 0, stream>>>(
          cur, plow, B, H, W, Dd, start);
    }

    // 4) upsample to full res (+ residual from previous scale)
    {
      long np = (long)B * HF * WF;
      const float* prev = (s == 0) ? nullptr : (out + (long)(s - 1) * FULLPIX);
      upsample_add_kernel<<<(unsigned)((np + 255) / 256), 256, 0, stream>>>(
          plow, H, W, prev, out + (long)s * FULLPIX, B);
    }
  }
}
